// CrosslayerAttentionModule_63050119905675
// MI455X (gfx1250) — matrix operand order repaired and checked by reading, not compiled, not run on hardware
//
#include <hip/hip_runtime.h>
#include <hip/hip_bf16.h>

typedef __attribute__((ext_vector_type(16))) _Float16 v16h;
typedef __attribute__((ext_vector_type(8)))  _Float16 v8h;
typedef __attribute__((ext_vector_type(8)))  float    v8f;
typedef __attribute__((ext_vector_type(4)))  int      v4i;
typedef __attribute__((address_space(1))) v4i v4i_glob;   // global (as1)
typedef __attribute__((address_space(3))) v4i v4i_lds;    // LDS    (as3)

#define DIM   256
#define NHEAD 8
#define HDIM  32
#define NB    4
#define NQ    4096   // 64*64 query pixels
#define NKV   1024   // 32*32 key pixels

#if defined(__has_builtin)
#if __has_builtin(__builtin_amdgcn_global_load_async_to_lds_b128)
#define USE_ASYNC_LDS 1
#endif
#endif

__device__ __forceinline__ v8f wmma_f16(v16h a, v16h b, v8f c) {
  // D = A(16x32 f16) * B(32x16 f16) + C(16x16 f32)
  return __builtin_amdgcn_wmma_f32_16x16x32_f16(false, a, false, b,
                                                (short)0, c, false, false);
}

// 16-byte HBM->LDS copy: async (ASYNCcnt-tracked, bypasses VGPRs) when the
// gfx1250 builtin exists, else synchronous vector copy.
__device__ __forceinline__ void copy16_to_lds(const _Float16* gsrc, _Float16* ldst) {
#ifdef USE_ASYNC_LDS
  __builtin_amdgcn_global_load_async_to_lds_b128(
      (v4i_glob*)gsrc, (v4i_lds*)ldst, 0, 0);
#else
  *(v8h*)ldst = *(const v8h*)gsrc;
#endif
}
__device__ __forceinline__ void copy_wait() {
#ifdef USE_ASYNC_LDS
#if __has_builtin(__builtin_amdgcn_s_wait_asynccnt)
  __builtin_amdgcn_s_wait_asynccnt(0);
#else
  asm volatile("s_wait_asynccnt 0x0" ::: "memory");
#endif
#endif
}

// ---------------------------------------------------------------------------
// 16-lane (DPP row) butterfly reductions. Masks {1,2,7,15} generate the full
// 16-lane group with disjoint cosets per step -> valid for max and sum.
// ---------------------------------------------------------------------------
template <int CTRL>
__device__ __forceinline__ float dpp_movf(float x) {
  return __int_as_float(__builtin_amdgcn_update_dpp(
      0, __float_as_int(x), CTRL, 0xF, 0xF, true));
}
__device__ __forceinline__ float row_max16(float m) {
  m = fmaxf(m, dpp_movf<0xB1>(m));   // quad_perm(1,0,3,2): xor 1
  m = fmaxf(m, dpp_movf<0x4E>(m));   // quad_perm(2,3,0,1): xor 2
  m = fmaxf(m, dpp_movf<0x141>(m));  // row_half_mirror:    xor 7
  m = fmaxf(m, dpp_movf<0x140>(m));  // row_mirror:         xor 15
  return m;
}
__device__ __forceinline__ float row_sum16(float s) {
  s += dpp_movf<0xB1>(s);
  s += dpp_movf<0x4E>(s);
  s += dpp_movf<0x141>(s);
  s += dpp_movf<0x140>(s);
  return s;
}

// ---------------------------------------------------------------------------
// f32 [B][C][Np] -> f16 [B][Np][C] tiled transpose+convert (32x32 LDS tile).
// ---------------------------------------------------------------------------
__global__ __launch_bounds__(256) void transpose_cvt_kernel(
    const float* __restrict__ in, _Float16* __restrict__ outT, int Np)
{
  __shared__ float tile[32][33];
  const int tid = threadIdx.x;
  const int p0 = blockIdx.x * 32, c0 = blockIdx.y * 32, b = blockIdx.z;
  const float* src = in + (size_t)b * DIM * Np;
#pragma unroll
  for (int i = 0; i < 4; ++i) {
    int r = (tid >> 5) + i * 8;
    tile[r][tid & 31] = src[(size_t)(c0 + r) * Np + p0 + (tid & 31)];
  }
  __syncthreads();
  _Float16* dst = outT + (size_t)b * Np * DIM;
#pragma unroll
  for (int i = 0; i < 4; ++i) {
    int pr = (tid >> 5) + i * 8;
    dst[(size_t)(p0 + pr) * DIM + c0 + (tid & 31)] = (_Float16)tile[tid & 31][pr];
  }
}

__global__ __launch_bounds__(256) void cvt_f16_kernel(
    const float* __restrict__ in, _Float16* __restrict__ out, int n)
{
  int i = blockIdx.x * 256 + threadIdx.x;
  if (i < n) out[i] = (_Float16)in[i];
}

// ---------------------------------------------------------------------------
// 1x1-conv GEMM, f16 operands, async double-buffered LDS staging.
// X is channel-last f16 [B][Np][DIM]; W is f16 [Otot][DIM].
// Block: 256 threads (8 waves, 2x4). Tile O=32 x P=128; each wave computes a
// 16x32 slab (1 A-frag, 2 B-frags, 2 WMMAs per 32-deep K step).
// OMODE 0: Q proj   -> f16 out_h0 [b][h][n][d]
// OMODE 1: KV proj  -> k: f16 out_h0 [b][h][nu][d]
//                      v: f16 out_h1 [b][h][d][nu]; f32 out_f [b][nu][c] (CL)
// OMODE 2: final    -> f32 out_f [b][c][n] (d_out)
// ---------------------------------------------------------------------------
template <int OMODE>
__global__ __launch_bounds__(256) void conv1x1_wmma_kernel(
    const _Float16* __restrict__ X,
    const _Float16* __restrict__ Wh,
    const float* __restrict__ scale,
    const float* __restrict__ bias,
    int Np,
    _Float16* __restrict__ out_h0,
    _Float16* __restrict__ out_h1,
    float* __restrict__ out_f)
{
  __shared__ __align__(16) _Float16 Wt[2][32][40];   // [o][k]
  __shared__ __align__(16) _Float16 Xt[2][128][40];  // [p][k]

  const int tid  = threadIdx.x;
  const int lane = tid & 31;
  const int wave = tid >> 5;
  const int wm = wave >> 2, wn = wave & 3;  // 2 (O) x 4 (P) waves
  const int lh = lane & 15, hl = lane >> 4;

  const int p0 = blockIdx.x * 128;
  const int o0 = blockIdx.y * 32;
  const int b  = blockIdx.z;

  const _Float16* Xb = X + (size_t)b * Np * DIM;

  // Stage one (W,X) K-tile into buffer `buf` with 16B copies.
  auto stage = [&](int buf, int k0) {
    if (tid < 128) {                      // 32 rows x 64B of W
      int r = tid >> 2, c8 = (tid & 3) * 8;
      copy16_to_lds(Wh + (size_t)(o0 + r) * DIM + k0 + c8, &Wt[buf][r][c8]);
    }
#pragma unroll
    for (int i = 0; i < 2; ++i) {         // 128 rows x 64B of X
      int idx = tid + i * 256;
      int p = idx >> 2, c8 = (idx & 3) * 8;
      copy16_to_lds(Xb + (size_t)(p0 + p) * DIM + k0 + c8, &Xt[buf][p][c8]);
    }
  };

  stage(0, 0);
  v8f acc0 = {}, acc1 = {};
  for (int k0 = 0, it = 0; k0 < DIM; k0 += 32, ++it) {
    const int cur = it & 1;
    copy_wait();            // this wave's copies for `cur` have landed
    __syncthreads();        // all waves' copies landed; prev compute finished
    if (k0 + 32 < DIM) stage(cur ^ 1, k0 + 32);   // overlap next tile w/ WMMA

    // A fragment (16x32): lane lh holds row wm*16+lh; K chunks per ISA layout.
    const int arow = wm * 16 + lh;
    const int kba  = hl ? 8 : 0;
    v8h alo = *(const v8h*)&Wt[cur][arow][kba];
    v8h ahi = *(const v8h*)&Wt[cur][arow][kba + 16];
    v16h a;
#pragma unroll
    for (int j = 0; j < 8; ++j) { a[j] = alo[j]; a[8 + j] = ahi[j]; }

    // Two B fragments (32x16): columns wn*32+lh and wn*32+16+lh.
    const int kbb = hl ? 8 : 0;
    const int br0 = wn * 32 + lh;
    v8h b00 = *(const v8h*)&Xt[cur][br0][kbb];
    v8h b01 = *(const v8h*)&Xt[cur][br0][kbb + 16];
    v8h b10 = *(const v8h*)&Xt[cur][br0 + 16][kbb];
    v8h b11 = *(const v8h*)&Xt[cur][br0 + 16][kbb + 16];
    v16h bf0, bf1;
#pragma unroll
    for (int j = 0; j < 8; ++j) {
      bf0[j] = b00[j]; bf0[8 + j] = b01[j];
      bf1[j] = b10[j]; bf1[8 + j] = b11[j];
    }

    acc0 = wmma_f16(a, bf0, acc0);
    acc1 = wmma_f16(a, bf1, acc1);
  }

  // Epilogue: C/D frag element i -> (M = i + hl*8, N = lh).
  const int pA = p0 + wn * 32 + lh;
  const int pB = pA + 16;
#pragma unroll
  for (int i = 0; i < 8; ++i) {
    const int o = o0 + wm * 16 + i + (hl ? 8 : 0);
    const float sc = scale[o], bi = bias[o];
    float vA = acc0[i] * sc + bi;
    float vB = acc1[i] * sc + bi;
    if (OMODE == 0) {
      int h = o >> 5, d = o & 31;
      _Float16* dst = out_h0 + ((size_t)b * NHEAD + h) * NQ * HDIM + d;
      dst[(size_t)pA * HDIM] = (_Float16)vA;
      dst[(size_t)pB * HDIM] = (_Float16)vB;
    } else if (OMODE == 1) {
      if (o < DIM) {
        int h = o >> 5, d = o & 31;
        _Float16* dst = out_h0 + ((size_t)b * NHEAD + h) * NKV * HDIM + d;
        dst[(size_t)pA * HDIM] = (_Float16)vA;
        dst[(size_t)pB * HDIM] = (_Float16)vB;
      } else {
        int oc = o - DIM;
        int h = oc >> 5, d = oc & 31;
        out_f[((size_t)b * NKV + pA) * DIM + oc] = vA;        // channel-last
        out_f[((size_t)b * NKV + pB) * DIM + oc] = vB;
        _Float16* dh = out_h1 + (((size_t)b * NHEAD + h) * HDIM + d) * NKV;
        dh[pA] = (_Float16)vA;
        dh[pB] = (_Float16)vB;
      }
    } else {
      float* df = out_f + ((size_t)b * DIM + o) * NQ;
      df[pA] = vA;
      df[pB] = vB;
    }
  }
}

// ---------------------------------------------------------------------------
// Flash attention: per (b,h), each wave owns 16 queries, streams 32 keys/iter.
// Running max via DPP16 butterflies; denominator kept as per-lane partials
// (identical alpha sequence across each 16-lane row group), reduced once at
// the end. Output written channel-last (coalesced 64B half-wave stores).
// ---------------------------------------------------------------------------
__global__ __launch_bounds__(256) void flash_attn_kernel(
    const _Float16* __restrict__ qh,   // [B*H][NQ][32]
    const _Float16* __restrict__ kh,   // [B*H][NKV][32]
    const _Float16* __restrict__ vhT,  // [B*H][32][NKV]
    float* __restrict__ ao)            // [B][NQ][256] channel-last
{
  __shared__ __align__(16) _Float16 Pt[8][16][32];  // per-wave P tile

  const int tid  = threadIdx.x;
  const int lane = tid & 31;
  const int wave = tid >> 5;
  const int lh = lane & 15, hl = lane >> 4;

  const int bh = blockIdx.y;           // b*8 + h
  const int b = bh >> 3, h = bh & 7;
  const int qbase = blockIdx.x * 128 + wave * 16;

  const _Float16* qrow = qh + ((size_t)bh * NQ + qbase + lh) * HDIM;
  const int kba = hl ? 16 : 0;
  v8h qlo = *(const v8h*)(qrow + kba);
  v8h qhi = *(const v8h*)(qrow + kba + 8);
  v16h af;
#pragma unroll
  for (int j = 0; j < 8; ++j) { af[j] = qlo[j]; af[8 + j] = qhi[j]; }

  const float sm_scale = 0.17677669529663687f;  // 1/sqrt(32)

  float row_max[8], row_sum[8];
  v8f o0 = {}, o1 = {};
#pragma unroll
  for (int i = 0; i < 8; ++i) { row_max[i] = -1e30f; row_sum[i] = 0.0f; }

  const _Float16* kbp = kh  + (size_t)bh * NKV * HDIM;
  const _Float16* vbp = vhT + (size_t)bh * HDIM * NKV;

  for (int kc = 0; kc < NKV; kc += 32) {
    const int kbd = hl ? 16 : 0;
    v16h bk0 = *(const v16h*)(kbp + (size_t)(kc + lh) * HDIM + kbd);
    v16h bk1 = *(const v16h*)(kbp + (size_t)(kc + 16 + lh) * HDIM + kbd);
    v8f s0 = {}, s1 = {};
    s0 = wmma_f16(af, bk0, s0);
    s1 = wmma_f16(af, bk1, s1);

#pragma unroll
    for (int i = 0; i < 8; ++i) {
      float x0 = s0[i] * sm_scale;
      float x1 = s1[i] * sm_scale;
      float m  = row_max16(fmaxf(x0, x1));
      float nm    = fmaxf(row_max[i], m);
      float alpha = __expf(row_max[i] - nm);
      row_max[i]  = nm;
      float p0v = __expf(x0 - nm);
      float p1v = __expf(x1 - nm);
      row_sum[i] = row_sum[i] * alpha + p0v + p1v;  // per-lane partial
      o0[i] *= alpha;
      o1[i] *= alpha;
      int r = i + (hl ? 8 : 0);
      Pt[wave][r][lh]      = (_Float16)p0v;
      Pt[wave][r][16 + lh] = (_Float16)p1v;
    }

    v8h plo = *(const v8h*)&Pt[wave][lh][kba];
    v8h phi = *(const v8h*)&Pt[wave][lh][kba + 8];
    v16h pf;
#pragma unroll
    for (int j = 0; j < 8; ++j) { pf[j] = plo[j]; pf[8 + j] = phi[j]; }

    const int vkb = kc + (hl ? 16 : 0);
    v16h bv0 = *(const v16h*)(vbp + (size_t)lh * NKV + vkb);
    v16h bv1 = *(const v16h*)(vbp + (size_t)(16 + lh) * NKV + vkb);
    o0 = wmma_f16(pf, bv0, o0);
    o1 = wmma_f16(pf, bv1, o1);
  }

#pragma unroll
  for (int i = 0; i < 8; ++i) {
    int   qr  = qbase + i + (hl ? 8 : 0);
    float inv = 1.0f / row_sum16(row_sum[i]);
    float* aorow = ao + ((size_t)b * NQ + qr) * DIM + h * HDIM;
    aorow[lh]      = o0[i] * inv;     // coalesced per 16-lane group
    aorow[16 + lh] = o1[i] * inv;
  }
}

// ---------------------------------------------------------------------------
// Depthwise 7x7 conv (pad 3) + scale/bias, channel-last (coalesced over c).
// ---------------------------------------------------------------------------
__global__ __launch_bounds__(256) void dwconv7_kernel(
    const float* __restrict__ vr,    // [B][1024][256]
    const float* __restrict__ Wpe,   // [256][49]
    const float* __restrict__ ps,
    const float* __restrict__ pb,
    float* __restrict__ vpe)         // [B][1024][256]
{
  const int p = blockIdx.x, b = blockIdx.y;
  const int c = threadIdx.x;
  const int py = p >> 5, px = p & 31;
  const float* src = vr + (size_t)b * NKV * DIM + c;
  float acc = 0.0f;
#pragma unroll
  for (int ky = 0; ky < 7; ++ky) {
    int y = py + ky - 3;
    if ((unsigned)y >= 32u) continue;
#pragma unroll
    for (int kx = 0; kx < 7; ++kx) {
      int x = px + kx - 3;
      if ((unsigned)x >= 32u) continue;
      acc += Wpe[c * 49 + ky * 7 + kx] * src[(size_t)(y * 32 + x) * DIM];
    }
  }
  vpe[((size_t)b * NKV + p) * DIM + c] = acc * ps[c] + pb[c];
}

// ---------------------------------------------------------------------------
// aoh = f16( ao + bilinear_upsample(vpe) ), all channel-last, coalesced.
// ---------------------------------------------------------------------------
__global__ __launch_bounds__(256) void upsample_add_kernel(
    const float* __restrict__ ao,    // [B][4096][256]
    const float* __restrict__ vpe,   // [B][1024][256]
    _Float16* __restrict__ aoh)      // [B][4096][256]
{
  size_t i = (size_t)blockIdx.x * 256 + threadIdx.x;
  const size_t total = (size_t)NB * NQ * DIM;
  if (i >= total) return;
  int    c  = (int)(i & (DIM - 1));
  size_t bp = i >> 8;
  int    p  = (int)(bp & (NQ - 1));
  int    b  = (int)(bp >> 12);
  int px = p & 63, py = p >> 6;
  float sx = px * 0.5f - 0.25f;
  float sy = py * 0.5f - 0.25f;
  int xf = (int)floorf(sx), yf = (int)floorf(sy);
  float wx = sx - xf, wy = sy - yf;
  int x0 = ::max(xf, 0),      y0 = ::max(yf, 0);
  int x1 = ::min(xf + 1, 31), y1 = ::min(yf + 1, 31);
  x0 = ::min(x0, 31); y0 = ::min(y0, 31);
  const float* s = vpe + ((size_t)b * NKV) * DIM + c;
  float v00 = s[(size_t)(y0 * 32 + x0) * DIM], v01 = s[(size_t)(y0 * 32 + x1) * DIM];
  float v10 = s[(size_t)(y1 * 32 + x0) * DIM], v11 = s[(size_t)(y1 * 32 + x1) * DIM];
  float v = (1.0f - wy) * ((1.0f - wx) * v00 + wx * v01)
          +         wy  * ((1.0f - wx) * v10 + wx * v11);
  aoh[i] = (_Float16)(ao[i] + v);
}

// ---------------------------------------------------------------------------
extern "C" void kernel_launch(void* const* d_in, const int* in_sizes, int n_in,
                              void* d_out, int out_size, void* d_ws, size_t ws_size,
                              hipStream_t stream) {
  (void)in_sizes; (void)n_in; (void)out_size; (void)ws_size;
  const float* x     = (const float*)d_in[0];
  const float* upper = (const float*)d_in[1];
  const float* Wq    = (const float*)d_in[2];
  const float* qs    = (const float*)d_in[3];
  const float* qb    = (const float*)d_in[4];
  const float* Wkv   = (const float*)d_in[5];
  const float* kvs   = (const float*)d_in[6];
  const float* kvb   = (const float*)d_in[7];
  const float* Wpe   = (const float*)d_in[8];
  const float* pes   = (const float*)d_in[9];
  const float* peb   = (const float*)d_in[10];
  const float* Wpr   = (const float*)d_in[11];
  const float* prs   = (const float*)d_in[12];
  const float* prb   = (const float*)d_in[13];
  float* out = (float*)d_out;

  char* w = (char*)d_ws;
  _Float16* xh   = (_Float16*)w; w += (size_t)NB * NQ  * DIM * sizeof(_Float16);  // 8 MB
  _Float16* uh   = (_Float16*)w; w += (size_t)NB * NKV * DIM * sizeof(_Float16);  // 2 MB
  _Float16* Wqh  = (_Float16*)w; w += (size_t)DIM * DIM * sizeof(_Float16);
  _Float16* Wkvh = (_Float16*)w; w += (size_t)2 * DIM * DIM * sizeof(_Float16);
  _Float16* Wprh = (_Float16*)w; w += (size_t)DIM * DIM * sizeof(_Float16);
  _Float16* qh   = (_Float16*)w; w += (size_t)NB * NHEAD * NQ  * HDIM * sizeof(_Float16); // 8 MB
  _Float16* kh   = (_Float16*)w; w += (size_t)NB * NHEAD * NKV * HDIM * sizeof(_Float16); // 2 MB
  _Float16* vhT  = (_Float16*)w; w += (size_t)NB * NHEAD * HDIM * NKV * sizeof(_Float16); // 2 MB
  float* vr  = (float*)w; w += (size_t)NB * NKV * DIM * sizeof(float);            // 4 MB
  float* vpe = (float*)w; w += (size_t)NB * NKV * DIM * sizeof(float);            // 4 MB
  float* ao  = (float*)w;                                                         // 16 MB
  _Float16* aoh = xh;   // xh is dead after the Q projection; reuse for aoh

  // f32 -> f16 staging (transpose activations to channel-last; weights flat).
  transpose_cvt_kernel<<<dim3(NQ / 32, DIM / 32, NB), 256, 0, stream>>>(x, xh, NQ);
  transpose_cvt_kernel<<<dim3(NKV / 32, DIM / 32, NB), 256, 0, stream>>>(upper, uh, NKV);
  cvt_f16_kernel<<<dim3(DIM * DIM / 256), 256, 0, stream>>>(Wq, Wqh, DIM * DIM);
  cvt_f16_kernel<<<dim3(2 * DIM * DIM / 256), 256, 0, stream>>>(Wkv, Wkvh, 2 * DIM * DIM);
  cvt_f16_kernel<<<dim3(DIM * DIM / 256), 256, 0, stream>>>(Wpr, Wprh, DIM * DIM);

  // Q projection: [256x256] x [256x4096] per batch.
  conv1x1_wmma_kernel<0><<<dim3(NQ / 128, DIM / 32, NB), 256, 0, stream>>>(
      xh, Wqh, qs, qb, NQ, qh, nullptr, nullptr);
  // KV projection: [512x256] x [256x1024] per batch.
  conv1x1_wmma_kernel<1><<<dim3(NKV / 128, (2 * DIM) / 32, NB), 256, 0, stream>>>(
      uh, Wkvh, kvs, kvb, NKV, kh, vhT, vr);
  // Depthwise 7x7 positional conv on v.
  dwconv7_kernel<<<dim3(NKV, NB), 256, 0, stream>>>(vr, Wpe, pes, peb, vpe);
  // Flash attention -> ao (channel-last).
  flash_attn_kernel<<<dim3(NQ / 128, NB * NHEAD), 256, 0, stream>>>(qh, kh, vhT, ao);
  // aoh = f16(ao + upsample(vpe)).
  upsample_add_kernel<<<dim3((unsigned)((size_t)NB * NQ * DIM / 256)), 256, 0, stream>>>(
      ao, vpe, aoh);
  // Final projection -> d_out.
  conv1x1_wmma_kernel<2><<<dim3(NQ / 128, DIM / 32, NB), 256, 0, stream>>>(
      aoh, Wprh, prs, prb, NQ, nullptr, nullptr, out);
}
